// BIRNN_encoder_decoder_9268539425250
// MI455X (gfx1250) — matrix operand order repaired and checked by reading, not compiled, not run on hardware
//
#include <hip/hip_runtime.h>
#include <math.h>

// ---------------------------------------------------------------------------
// BIRNN encoder-decoder for MI455X (gfx1250, wave32, WMMA).
// All 512/1024-wide GEMM steps run on v_wmma_f32_16x16x32_f16.
// Weights are converted to f16 once per launch (they are tiny: <8MB, L2
// resident on the 192MB L2), hidden state is kept in both f32 (for the final
// decoder dot / numerics) and f16 (as WMMA A operand for the next step).
// Recurrence is sequenced by per-step kernel launches on `stream`.
// ---------------------------------------------------------------------------

typedef _Float16 half8 __attribute__((ext_vector_type(8)));
typedef _Float16 v16h  __attribute__((ext_vector_type(16)));
typedef float    v8f   __attribute__((ext_vector_type(8)));

#define BB    128   // batch
#define TT    128   // encoder sequence length
#define INW   3     // raw input width
#define HH    512   // hidden
#define TGT_N 32    // decoder rollout length

// ---------------------------------------------------------------- utilities
__global__ void f32_to_f16_kernel(const float* __restrict__ src,
                                  _Float16* __restrict__ dst, int n) {
  int i = blockIdx.x * blockDim.x + threadIdx.x;
  int stride = gridDim.x * blockDim.x;
  for (; i < n; i += stride) dst[i] = (_Float16)src[i];
}

__global__ void zero16_kernel(_Float16* __restrict__ dst, int n) {
  int i = blockIdx.x * blockDim.x + threadIdx.x;
  int stride = gridDim.x * blockDim.x;
  for (; i < n; i += stride) dst[i] = (_Float16)0.0f;
}

// --------------------------------------------------- WMMA fragment loaders
// A fragment (16x32, f16): lane L (M = L&15) holds K = kb..kb+7 and
// kb+16..kb+23 where kb = (L<16 ? 0 : 8).  Two contiguous 16B loads.
__device__ __forceinline__ v16h load_a_frag(const _Float16* rowAtK, int lane) {
  const int kb = (lane & 16) ? 8 : 0;
  half8 lo = *reinterpret_cast<const half8*>(rowAtK + kb);
  half8 hi = *reinterpret_cast<const half8*>(rowAtK + kb + 16);
  return __builtin_shufflevector(lo, hi, 0,1,2,3,4,5,6,7,8,9,10,11,12,13,14,15);
}

// B fragment (32x16, f16), B[k][n] = W[h0+n][k] with W row-major (HxK):
// lane L holds column n = L&15, K = kb..kb+7 and kb+16..kb+23 where kb = (L<16 ? 0 : 8).
// two runs of 8 contiguous halves along the W row -> two 16B loads.
__device__ __forceinline__ v16h load_b_frag(const _Float16* wRowAtK, int lane) {
  const int kb = (lane & 16) ? 8 : 0;
  half8 lo = *reinterpret_cast<const half8*>(wRowAtK + kb);
  half8 hi = *reinterpret_cast<const half8*>(wRowAtK + kb + 16);
  return __builtin_shufflevector(lo, hi, 0,1,2,3,4,5,6,7,8,9,10,11,12,13,14,15);
}

// ------------------------------------------------------- generic RNN step
// h_new = tanh( A0 @ W0^T  +  X3 @ W3^T  +  Hprev @ Whh^T  +  bih + bhh )
//   A0 : (BB x K0) f16, row stride a0_stride          (optional, K0==0 -> off)
//   X3 : (BB x 3)  f32, row stride x3_stride          (optional, nullptr)
//   Hprev : (BB x HH) f16 ; Whh : (HH x HH) f16
// Grid: 64 blocks x 128 threads = 256 waves = (BB/16) x (HH/16) tiles.
__global__ __launch_bounds__(128)
void rnn_step_wmma(const _Float16* __restrict__ A0,
                   const _Float16* __restrict__ W0, int K0, int a0_stride,
                   const float* __restrict__ X3,
                   const float* __restrict__ W3, int x3_stride,
                   const _Float16* __restrict__ Hprev,
                   const _Float16* __restrict__ Whh,
                   const float* __restrict__ bih,
                   const float* __restrict__ bhh,
                   float* __restrict__ Hout32,
                   _Float16* __restrict__ Hout16,
                   _Float16* __restrict__ SeqOut, int seq_stride) {
  const int wave = (blockIdx.x * blockDim.x + threadIdx.x) >> 5;
  const int lane = threadIdx.x & 31;
  const int bt = wave >> 5;                 // 0..7  batch tile
  const int ht = wave & 31;                 // 0..31 hidden tile
  const int b0 = bt * 16, h0 = ht * 16;
  const int arow = b0 + (lane & 15);        // A-operand source row (batch)
  const int wrow = h0 + (lane & 15);        // W row feeding B-operand column

  v8f acc = {};

  if (K0 > 0) {  // fused input projection (layer-1 encoder / decoder l>0)
    const _Float16* aBase = A0 + (size_t)arow * a0_stride;
    const _Float16* wBase = W0 + (size_t)wrow * K0;
#pragma unroll 4
    for (int k = 0; k < K0; k += 32) {
      v16h a = load_a_frag(aBase + k, lane);
      v16h b = load_b_frag(wBase + k, lane);
      acc = __builtin_amdgcn_wmma_f32_16x16x32_f16(false, a, false, b,
                                                   (short)0, acc, false, false);
    }
  }
  {  // recurrent term, K = HH
    const _Float16* aBase = Hprev + (size_t)arow * HH;
    const _Float16* wBase = Whh + (size_t)wrow * HH;
#pragma unroll 4
    for (int k = 0; k < HH; k += 32) {
      v16h a = load_a_frag(aBase + k, lane);
      v16h b = load_b_frag(wBase + k, lane);
      acc = __builtin_amdgcn_wmma_f32_16x16x32_f16(false, a, false, b,
                                                   (short)0, acc, false, false);
    }
  }

  // Epilogue. C/D layout: VGPR r, lanes 0-15 -> M=r, lanes 16-31 -> M=r+8;
  // column N = lane & 15.
  const int n = lane & 15;
  const int h = h0 + n;
  const float bsum = bih[h] + bhh[h];
  float w30 = 0.f, w31 = 0.f, w32 = 0.f;
  if (X3) { w30 = W3[h * 3 + 0]; w31 = W3[h * 3 + 1]; w32 = W3[h * 3 + 2]; }

#pragma unroll
  for (int r = 0; r < 8; ++r) {
    const int m = r + ((lane & 16) ? 8 : 0);
    const int b = b0 + m;
    float v = acc[r] + bsum;
    if (X3) {
      const float* xr = X3 + (size_t)b * x3_stride;
      v += xr[0] * w30 + xr[1] * w31 + xr[2] * w32;
    }
    v = tanhf(v);
    Hout32[(size_t)b * HH + h] = v;
    Hout16[(size_t)b * HH + h] = (_Float16)v;
    if (SeqOut) SeqOut[(size_t)b * seq_stride + h] = (_Float16)v;
  }
}

// ------------------------------------------------ decoder output/feedback
__global__ void dec_out_step(const float* __restrict__ h3,     // (BB x HH) f32
                             const float* __restrict__ linW,   // (HH)
                             const float* __restrict__ linb,   // (1)
                             const float* __restrict__ xin, int xin_stride,
                             float* __restrict__ nxt,          // (BB x 3)
                             float* __restrict__ out, int step) {
  const int b = blockIdx.x * blockDim.x + threadIdx.x;
  if (b >= BB) return;
  const float* hr = h3 + (size_t)b * HH;
  float o = linb[0];
#pragma unroll 8
  for (int k = 0; k < HH; ++k) o += hr[k] * linW[k];
  out[(size_t)b * TGT_N + step] = o;
  const float* xr = xin + (size_t)b * xin_stride;
  const float s1 = xr[0] - o;
  const float s2 = xr[1] - s1;
  nxt[b * 3 + 0] = o;
  nxt[b * 3 + 1] = s1;
  nxt[b * 3 + 2] = s2;
}

// ---------------------------------------------------------------- launcher
extern "C" void kernel_launch(void* const* d_in, const int* in_sizes, int n_in,
                              void* d_out, int out_size, void* d_ws,
                              size_t ws_size, hipStream_t stream) {
  (void)in_sizes; (void)n_in; (void)out_size; (void)ws_size;
  const float* x        = (const float*)d_in[0];
  const float* enc_Wih0 = (const float*)d_in[2];   // (2,H,3)
  const float* enc_Whh0 = (const float*)d_in[3];   // (2,H,H)
  const float* enc_Wih1 = (const float*)d_in[4];   // (2,H,2H)
  const float* enc_Whh1 = (const float*)d_in[5];   // (2,H,H)
  const float* enc_bih  = (const float*)d_in[6];   // (2,2,H)
  const float* enc_bhh  = (const float*)d_in[7];   // (2,2,H)
  const float* dec_Wih0 = (const float*)d_in[8];   // (H,3)
  const float* dec_Wihr = (const float*)d_in[9];   // (3,H,H)
  const float* dec_Whh  = (const float*)d_in[10];  // (4,H,H)
  const float* dec_bih  = (const float*)d_in[11];  // (4,H)
  const float* dec_bhh  = (const float*)d_in[12];  // (4,H)
  const float* lin_W    = (const float*)d_in[13];  // (1,H)
  const float* lin_b    = (const float*)d_in[14];  // (1,)
  float* out = (float*)d_out;

  // ---- workspace carve-up (256B aligned slabs) ----
  size_t off = 0;
  auto alloc = [&](size_t bytes) -> char* {
    char* p = (char*)d_ws + off;
    off += (bytes + 255) & ~(size_t)255;
    return p;
  };
  _Float16* wWhh0 = (_Float16*)alloc((size_t)2 * HH * HH * 2);
  _Float16* wWhh1 = (_Float16*)alloc((size_t)2 * HH * HH * 2);
  _Float16* wWih1 = (_Float16*)alloc((size_t)2 * HH * 2 * HH * 2);
  _Float16* wWihr = (_Float16*)alloc((size_t)3 * HH * HH * 2);
  _Float16* wDWhh = (_Float16*)alloc((size_t)4 * HH * HH * 2);
  _Float16* l0seq = (_Float16*)alloc((size_t)TT * BB * 2 * HH * 2);  // 32MB
  _Float16* h16[4][2];
  float*    h32[4][2];
  for (int l = 0; l < 4; ++l)
    for (int p = 0; p < 2; ++p) {
      h16[l][p] = (_Float16*)alloc((size_t)BB * HH * 2);
      h32[l][p] = (float*)alloc((size_t)BB * HH * 4);
    }
  float* decin[2] = {(float*)alloc(BB * 3 * 4), (float*)alloc(BB * 3 * 4)};

  // ---- weight conversion (per-launch, deterministic) ----
  auto cvt = [&](const float* s, _Float16* d, int n) {
    f32_to_f16_kernel<<<256, 256, 0, stream>>>(s, d, n);
  };
  cvt(enc_Whh0, wWhh0, 2 * HH * HH);
  cvt(enc_Whh1, wWhh1, 2 * HH * HH);
  cvt(enc_Wih1, wWih1, 2 * HH * 2 * HH);
  cvt(dec_Wihr, wWihr, 3 * HH * HH);
  cvt(dec_Whh,  wDWhh, 4 * HH * HH);

  const dim3 sgrid(64), sblk(128);  // 256 waves = 8x32 output tiles

  // ---- encoder scans: scan 0..3 = (L0 fwd, L0 bwd, L1 fwd, L1 bwd) ----
  int par[4];
  auto enc_scan = [&](int si, int layer, int dir) {
    int cur = 0;
    zero16_kernel<<<64, 256, 0, stream>>>(h16[si][cur], BB * HH);
    for (int s = 0; s < TT; ++s) {
      const int t = dir ? (TT - 1 - s) : s;
      const int nx = cur ^ 1;
      if (layer == 0) {
        rnn_step_wmma<<<sgrid, sblk, 0, stream>>>(
            nullptr, nullptr, 0, 0,
            x + (size_t)t * INW, enc_Wih0 + (size_t)dir * HH * INW, TT * INW,
            h16[si][cur], wWhh0 + (size_t)dir * HH * HH,
            enc_bih + (size_t)(0 * 2 + dir) * HH,
            enc_bhh + (size_t)(0 * 2 + dir) * HH,
            h32[si][nx], h16[si][nx],
            l0seq + (size_t)t * BB * (2 * HH) + (size_t)dir * HH, 2 * HH);
      } else {
        rnn_step_wmma<<<sgrid, sblk, 0, stream>>>(
            l0seq + (size_t)t * BB * (2 * HH),
            wWih1 + (size_t)dir * HH * (2 * HH), 2 * HH, 2 * HH,
            nullptr, nullptr, 0,
            h16[si][cur], wWhh1 + (size_t)dir * HH * HH,
            enc_bih + (size_t)(1 * 2 + dir) * HH,
            enc_bhh + (size_t)(1 * 2 + dir) * HH,
            h32[si][nx], h16[si][nx], nullptr, 0);
      }
      cur = nx;
    }
    par[si] = cur;
  };
  enc_scan(0, 0, 0);
  enc_scan(1, 0, 1);
  enc_scan(2, 1, 0);
  enc_scan(3, 1, 1);

  // ---- decoder: hn = [hf0, hb0, hf1, hb1] seed layers 0..3 ----
  int cur[4] = {par[0], par[1], par[2], par[3]};
  int dp = 0;
  for (int s = 0; s < TGT_N; ++s) {
    const float* xin = (s == 0) ? (x + (size_t)(TT - 1) * INW) : decin[dp];
    const int xstride = (s == 0) ? TT * INW : INW;
    for (int l = 0; l < 4; ++l) {
      const int nx = cur[l] ^ 1;
      if (l == 0) {
        rnn_step_wmma<<<sgrid, sblk, 0, stream>>>(
            nullptr, nullptr, 0, 0,
            xin, dec_Wih0, xstride,
            h16[0][cur[0]], wDWhh,
            dec_bih, dec_bhh,
            h32[0][nx], h16[0][nx], nullptr, 0);
      } else {
        rnn_step_wmma<<<sgrid, sblk, 0, stream>>>(
            h16[l - 1][cur[l - 1]], wWihr + (size_t)(l - 1) * HH * HH, HH, HH,
            nullptr, nullptr, 0,
            h16[l][cur[l]], wDWhh + (size_t)l * HH * HH,
            dec_bih + (size_t)l * HH, dec_bhh + (size_t)l * HH,
            h32[l][nx], h16[l][nx], nullptr, 0);
      }
      cur[l] = nx;
    }
    dec_out_step<<<1, 128, 0, stream>>>(h32[3][cur[3]], lin_W, lin_b,
                                        xin, xstride, decin[dp ^ 1], out, s);
    dp ^= 1;
  }
}
